// DPWFSTQuantizer_74594991997750
// MI455X (gfx1250) — matrix operand order repaired and checked by reading, not compiled, not run on hardware
//
#include <hip/hip_runtime.h>

#define T_N   16384
#define D_N   768
#define C_N   1024
#define K_N   32
#define KSTEPS (D_N / 32)      // 24 k-steps of 32
#define CT_N  (C_N / 16)       // 64 column tiles
#define NT    4                // column tiles per wave per pass (N-blocking)

typedef __attribute__((ext_vector_type(16))) __bf16 v16bf;
typedef __attribute__((ext_vector_type(8)))  float  v8f;

union ABReg { unsigned u[8]; v16bf v; uint4 q[2]; };

__device__ __forceinline__ int lane_id() {
  return (int)__builtin_amdgcn_mbcnt_lo(~0u, 0u);   // wave32: lane 0..31
}
__device__ __forceinline__ int xshfl_i(int v, int m) {
  int l = lane_id();
  return __builtin_amdgcn_ds_bpermute(((l ^ m) << 2), v);
}
__device__ __forceinline__ float xshfl_f(float v, int m) {
  int r = xshfl_i(__builtin_bit_cast(int, v), m);
  return __builtin_bit_cast(float, r);
}
__device__ __forceinline__ unsigned short f32_to_bf16_rne(float f) {
  unsigned u = __builtin_bit_cast(unsigned, f);
  unsigned r = u + 0x7fffu + ((u >> 16) & 1u);
  return (unsigned short)(r >> 16);
}
__device__ __forceinline__ float bf16_to_f32(unsigned short h) {
  unsigned u = ((unsigned)h) << 16;
  return __builtin_bit_cast(float, u);
}

// ---------------------------------------------------------------------------
// Kernel 1: split codebook into bf16 hi/lo, pre-swizzled into WMMA B layout.
// B (bf16, 32x16), k order shared with A: lanes 0-15 -> K=0..15 (VGPR r: K=2r,2r+1), lanes 16-31 -> K=16..31.
// Storage: dword index ((ct*KSTEPS + ks)*32 + lane)*8 + r  (fully coalesced loads).
// ---------------------------------------------------------------------------
__global__ void prep_cb_kernel(const float* __restrict__ cb,
                               unsigned* __restrict__ cbB_hi,
                               unsigned* __restrict__ cbB_lo) {
  int ct = blockIdx.x / KSTEPS;
  int ks = blockIdx.x % KSTEPS;
  int tid = threadIdx.x;          // 256 threads
  int lane = tid >> 3;            // 0..31
  int r = tid & 7;                // 0..7
  int n = ct * 16 + (lane & 15);
  int k = ks * 32 + ((lane >> 4) * 16) + 2 * r;
  float f0 = cb[n * D_N + k];
  float f1 = cb[n * D_N + k + 1];
  unsigned short h0 = f32_to_bf16_rne(f0);
  unsigned short h1 = f32_to_bf16_rne(f1);
  unsigned short l0 = f32_to_bf16_rne(f0 - bf16_to_f32(h0));
  unsigned short l1 = f32_to_bf16_rne(f1 - bf16_to_f32(h1));
  int o = ((ct * KSTEPS + ks) * 32 + lane) * 8 + r;
  cbB_hi[o] = ((unsigned)h1 << 16) | h0;
  cbB_lo[o] = ((unsigned)l1 << 16) | l0;
}

// ---------------------------------------------------------------------------
// Kernel 2: c2[n] = sum_k cb[n][k]^2
// ---------------------------------------------------------------------------
__global__ void c2_kernel(const float* __restrict__ cb, float* __restrict__ c2) {
  __shared__ float red[256];
  int n = blockIdx.x;
  float s = 0.f;
  for (int k = threadIdx.x; k < D_N; k += 256) {
    float v = cb[n * D_N + k];
    s += v * v;
  }
  red[threadIdx.x] = s;
  __syncthreads();
  for (int off = 128; off >= 1; off >>= 1) {
    if (threadIdx.x < off) red[threadIdx.x] += red[threadIdx.x + off];
    __syncthreads();
  }
  if (threadIdx.x == 0) c2[n] = red[0];
}

// ---------------------------------------------------------------------------
// Kernel 3: fused distances (split-bf16 WMMA GEMM) + stable top-32 per row.
// One block = 16 feature rows; 4 waves. N-blocking: each wave owns 4 adjacent
// 16-col tiles per pass (4 accumulators), so the A tile LDS loads are
// amortized over 12 WMMAs instead of 3.
// ---------------------------------------------------------------------------
__global__ void __launch_bounds__(128)
dist_topk_kernel(const float* __restrict__ feat,
                 const unsigned* __restrict__ cbB_hi,
                 const unsigned* __restrict__ cbB_lo,
                 const float* __restrict__ c2,
                 float* __restrict__ dist_top,
                 int* __restrict__ idx_top) {
  __shared__ unsigned short fhi[16 * D_N];   // 24 KB
  __shared__ unsigned short flo[16 * D_N];   // 24 KB
  __shared__ float distb[16 * C_N];          // 64 KB
  __shared__ float f2r[16];

  const int tid = threadIdx.x;
  const int row0 = blockIdx.x * 16;

  if (tid < 16) f2r[tid] = 0.f;
  __syncthreads();

  // Stage features: 8 threads per row, split to bf16 hi/lo, accumulate |f|^2.
  {
    int m = tid >> 3;         // row 0..15
    int sub = tid & 7;
    float ss = 0.f;
    for (int k = sub; k < D_N; k += 8) {
      float f = feat[(row0 + m) * D_N + k];
      unsigned short h = f32_to_bf16_rne(f);
      unsigned short l = f32_to_bf16_rne(f - bf16_to_f32(h));
      fhi[m * D_N + k] = h;
      flo[m * D_N + k] = l;
      ss += f * f;
    }
    atomicAdd(&f2r[m], ss);
  }
  __syncthreads();

  const int wave = tid >> 5;
  const int lane = tid & 31;
  const int mlo = lane & 15;
  const int half = lane >> 4;
  const unsigned* fhi32 = (const unsigned*)fhi;
  const unsigned* flo32 = (const unsigned*)flo;
  const uint4* bhi4 = (const uint4*)cbB_hi;
  const uint4* blo4 = (const uint4*)cbB_lo;

  for (int pass = 0; pass < CT_N / (4 * NT); ++pass) {
    int ct0 = (pass * 4 + wave) * NT;           // 4 adjacent column tiles
    v8f acc[NT];
#pragma unroll
    for (int tt = 0; tt < NT; ++tt) acc[tt] = (v8f){};

    for (int ks = 0; ks < KSTEPS; ++ks) {
      // A tile (16x32 bf16): lane m=lane&15; half selects K=16*half..+15. Loaded once,
      // reused across NT column tiles.
      ABReg ahi, alo;
#pragma unroll
      for (int r = 0; r < 8; ++r) {
        int k = ks * 32 + half * 16 + 2 * r;
        int di = (mlo * D_N + k) >> 1;
        ahi.u[r] = fhi32[di];
        alo.u[r] = flo32[di];
      }
#pragma unroll
      for (int tt = 0; tt < NT; ++tt) {
        ABReg bhi, blo;
        int qi = (((ct0 + tt) * KSTEPS + ks) * 32 + lane) * 2;
        bhi.q[0] = bhi4[qi];     bhi.q[1] = bhi4[qi + 1];
        blo.q[0] = blo4[qi];     blo.q[1] = blo4[qi + 1];
        acc[tt] = __builtin_amdgcn_wmma_f32_16x16x32_bf16(false, ahi.v, false, bhi.v,
                                                          (short)0, acc[tt], false, false);
        acc[tt] = __builtin_amdgcn_wmma_f32_16x16x32_bf16(false, ahi.v, false, blo.v,
                                                          (short)0, acc[tt], false, false);
        acc[tt] = __builtin_amdgcn_wmma_f32_16x16x32_bf16(false, alo.v, false, bhi.v,
                                                          (short)0, acc[tt], false, false);
      }
    }
    // finalize d2 = f2 + c2 - 2*dot, clamp 0; C/D layout: VGPR r -> M = r + 8*half.
#pragma unroll
    for (int tt = 0; tt < NT; ++tt) {
      int n0 = (ct0 + tt) * 16;
      float c2v = c2[n0 + mlo];
#pragma unroll
      for (int r = 0; r < 8; ++r) {
        int m = r + half * 8;
        float d = f2r[m] + c2v - 2.f * acc[tt][r];
        distb[m * C_N + n0 + mlo] = fmaxf(d, 0.f);
      }
    }
  }
  __syncthreads();

  // Stable top-32 per row: 32x (global min, smallest index wins ties).
  for (int rr = 0; rr < 4; ++rr) {
    int m = wave * 4 + rr;
    int grow = row0 + m;
    for (int sel = 0; sel < K_N; ++sel) {
      float bv = 3.0e38f;
      int bi = 0x7fffffff;
      for (int j = 0; j < C_N / 32; ++j) {
        int idx = j * 32 + lane;
        float v = distb[m * C_N + idx];
        if (v < bv) { bv = v; bi = idx; }     // ascending idx -> stable
      }
#pragma unroll
      for (int off = 16; off >= 1; off >>= 1) {
        float ov = xshfl_f(bv, off);
        int   oi = xshfl_i(bi, off);
        if (ov < bv || (ov == bv && oi < bi)) { bv = ov; bi = oi; }
      }
      if (lane == 0) {
        dist_top[grow * K_N + sel] = bv;
        idx_top[grow * K_N + sel] = bi;
        distb[m * C_N + bi] = 3.0e38f;        // knock out winner
      }
    }
  }
}

// ---------------------------------------------------------------------------
// Kernel 4: Viterbi forward DP. One wave32, lane = state j. Exact first-argmax
// semantics (strict >, 4 unrolled chains merged preferring smaller i).
// ---------------------------------------------------------------------------
__global__ void viterbi_kernel(const float* __restrict__ dist_top,
                               const int* __restrict__ idx_top,
                               const float* __restrict__ lmbda,
                               unsigned char* __restrict__ bp_g,
                               float* __restrict__ alphaT) {
  int j = threadIdx.x;            // 0..31
  float lam = lmbda[0];
  float alpha = -dist_top[j];
  int ip = idx_top[j];

  for (int t = 1; t < T_N; ++t) {
    int ic = idx_top[t * K_N + j];
    float dt = dist_top[t * K_N + j];
    float b0 = -3.0e38f, b1 = -3.0e38f, b2 = -3.0e38f, b3 = -3.0e38f;
    int p0 = 0, p1 = 1, p2 = 2, p3 = 3;
    int abits = __builtin_bit_cast(int, alpha);
#pragma unroll 4
    for (int i = 0; i < 32; i += 4) {
      float a0 = __builtin_bit_cast(float, __builtin_amdgcn_readlane(abits, i));
      float a1 = __builtin_bit_cast(float, __builtin_amdgcn_readlane(abits, i + 1));
      float a2 = __builtin_bit_cast(float, __builtin_amdgcn_readlane(abits, i + 2));
      float a3 = __builtin_bit_cast(float, __builtin_amdgcn_readlane(abits, i + 3));
      int x0 = __builtin_amdgcn_readlane(ip, i);
      int x1 = __builtin_amdgcn_readlane(ip, i + 1);
      int x2 = __builtin_amdgcn_readlane(ip, i + 2);
      int x3 = __builtin_amdgcn_readlane(ip, i + 3);
      float c0 = a0 + ((ic == x0) ? lam : 0.f);
      float c1 = a1 + ((ic == x1) ? lam : 0.f);
      float c2v = a2 + ((ic == x2) ? lam : 0.f);
      float c3 = a3 + ((ic == x3) ? lam : 0.f);
      if (c0 > b0) { b0 = c0; p0 = i; }
      if (c1 > b1) { b1 = c1; p1 = i + 1; }
      if (c2v > b2) { b2 = c2v; p2 = i + 2; }
      if (c3 > b3) { b3 = c3; p3 = i + 3; }
    }
    float best = b0; int bp = p0;
    if (b1 > best || (b1 == best && p1 < bp)) { best = b1; bp = p1; }
    if (b2 > best || (b2 == best && p2 < bp)) { best = b2; bp = p2; }
    if (b3 > best || (b3 == best && p3 < bp)) { best = b3; bp = p3; }
    alpha = best - dt;
    bp_g[t * K_N + j] = (unsigned char)bp;
    ip = ic;
  }
  alphaT[j] = alpha;
}

// ---------------------------------------------------------------------------
// Kernel 5: backtrack. Stage 4096-row bp chunks into LDS (coalesced), then a
// single lane walks the chain at LDS latency instead of L2 latency.
// ---------------------------------------------------------------------------
#define CHUNK 4096
__global__ void backtrack_kernel(const float* __restrict__ alphaT,
                                 const unsigned char* __restrict__ bp_g,
                                 int* __restrict__ path) {
  __shared__ unsigned bpw[CHUNK * 8];   // 128 KB
  __shared__ int s_cur;
  int tid = threadIdx.x;

  if (tid < 32) {
    float bv = alphaT[tid];
    int bi = tid;
#pragma unroll
    for (int off = 16; off >= 1; off >>= 1) {
      float ov = xshfl_f(bv, off);
      int   oi = xshfl_i(bi, off);
      if (ov > bv || (ov == bv && oi < bi)) { bv = ov; bi = oi; }   // first argmax
    }
    if (tid == 0) { s_cur = bi; path[T_N - 1] = bi; }
  }
  __syncthreads();

  const unsigned* bp32 = (const unsigned*)bp_g;
  for (int c = T_N / CHUNK - 1; c >= 0; --c) {
    int lo = c * CHUNK;
    for (int i = tid; i < CHUNK * 8; i += 256)
      bpw[i] = bp32[lo * 8 + i];
    __syncthreads();
    if (tid == 0) {
      int cur = s_cur;
      const unsigned char* bpb = (const unsigned char*)bpw;
      int tlo = (lo > 1) ? lo : 1;
      for (int t = lo + CHUNK - 1; t >= tlo; --t) {
        cur = (int)bpb[(t - lo) * K_N + cur];
        path[t - 1] = cur;
      }
      s_cur = cur;
    }
    __syncthreads();
  }
}

// ---------------------------------------------------------------------------
// Kernel 6: labels + dedup (repeat -> -1), parallel over T.
// ---------------------------------------------------------------------------
__global__ void emit_kernel(const int* __restrict__ idx_top,
                            const int* __restrict__ path,
                            int* __restrict__ out) {
  int t = blockIdx.x * blockDim.x + threadIdx.x;
  if (t >= T_N) return;
  int lab = idx_top[t * K_N + path[t]];
  int unit = lab;
  if (t > 0) {
    int prev = idx_top[(t - 1) * K_N + path[t - 1]];
    if (lab == prev) unit = -1;
  }
  out[t] = unit;
}

// ---------------------------------------------------------------------------
extern "C" void kernel_launch(void* const* d_in, const int* in_sizes, int n_in,
                              void* d_out, int out_size, void* d_ws, size_t ws_size,
                              hipStream_t stream) {
  const float* feat = (const float*)d_in[0];
  const float* cb   = (const float*)d_in[1];
  const float* lam  = (const float*)d_in[2];
  (void)in_sizes; (void)n_in; (void)ws_size; (void)out_size;

  char* ws = (char*)d_ws;
  size_t o = 0;
  auto alloc = [&](size_t bytes) { void* p = ws + o; o = (o + bytes + 255) & ~(size_t)255; return p; };

  unsigned* cbB_hi = (unsigned*)alloc((size_t)CT_N * KSTEPS * 32 * 8 * 4);   // 1.5 MB
  unsigned* cbB_lo = (unsigned*)alloc((size_t)CT_N * KSTEPS * 32 * 8 * 4);   // 1.5 MB
  float*    c2     = (float*)alloc(C_N * 4);
  float*    dtop   = (float*)alloc((size_t)T_N * K_N * 4);                   // 2 MB
  int*      itop   = (int*)alloc((size_t)T_N * K_N * 4);                     // 2 MB
  unsigned char* bp = (unsigned char*)alloc((size_t)T_N * K_N);              // 512 KB
  float*    alphaT = (float*)alloc(K_N * 4);
  int*      path   = (int*)alloc(T_N * 4);

  prep_cb_kernel<<<CT_N * KSTEPS, 256, 0, stream>>>(cb, cbB_hi, cbB_lo);
  c2_kernel<<<C_N, 256, 0, stream>>>(cb, c2);
  dist_topk_kernel<<<T_N / 16, 128, 0, stream>>>(feat, cbB_hi, cbB_lo, c2, dtop, itop);
  viterbi_kernel<<<1, 32, 0, stream>>>(dtop, itop, lam, bp, alphaT);
  backtrack_kernel<<<1, 256, 0, stream>>>(alphaT, bp, path);
  emit_kernel<<<T_N / 256, 256, 0, stream>>>(itop, path, (int*)d_out);
}
